// AESUELOGIT_13941463843222
// MI455X (gfx1250) — matrix operand order repaired and checked by reading, not compiled, not run on hardware
//
#include <hip/hip_runtime.h>
#include <hip/hip_bf16.h>
#include <stddef.h>

// Problem dims (from the reference)
#define N_DH   96        // N_DAYS * N_HOURS = 4*24
#define N_L    2000      // links
#define N_LP   2048      // links padded to multiple of 32 (WMMA K)
#define N_P    20000     // paths (multiple of 32)
#define N_ODS  5000
#define KSPLIT 5         // split-K factor for GEMM2 (625 k-steps / 5 = 125)

typedef __attribute__((ext_vector_type(16))) __bf16 bf16x16;
typedef __attribute__((ext_vector_type(8)))  __bf16 bf16x8;
typedef __attribute__((ext_vector_type(8)))  float  f32x8;
typedef __attribute__((ext_vector_type(4)))  float  f32x4;

// Workspace layout (bytes)
#define OFF_VBF   0                              // 96*2048 bf16      = 393,216 B
#define OFF_FBF   (OFF_VBF + N_DH*N_LP*2)        // 96*20000 bf16     = 3,840,000 B
#define OFF_VF    (OFF_FBF + N_DH*N_P*2)         // 96*20000 f32      = 7,680,000 B
#define OFF_PART  (OFF_VF  + (size_t)N_DH*N_P*4) // KSPLIT*96*2000 f32 = 3,840,000 B

// ---------------------------------------------------------------------------
// 16-bit WMMA A-fragment loader (contiguous K order, matches B fills):
// lanes 0-15 hold K = [0,16);
// lanes 16-31 hold K = [16,32).
// Two 16B loads + register concatenation (no extra VALU).
// ---------------------------------------------------------------------------
__device__ __forceinline__ bf16x16 load_a_frag(const __bf16* __restrict__ row,
                                               int k0, int kh) {
  const bf16x8 lo = *(const bf16x8*)(row + k0 + kh * 16);
  const bf16x8 hi = *(const bf16x8*)(row + k0 + 8 + kh * 16);
  return __builtin_shufflevector(lo, hi, 0, 1, 2, 3, 4, 5, 6, 7,
                                         8, 9, 10, 11, 12, 13, 14, 15);
}

// ---------------------------------------------------------------------------
// Kernel 1: V[dh][l] = X[dh][l][1..4]·theta + theta_links[l]; store bf16,
// zero-padded to N_LP columns (WMMA A operand for GEMM1).
// ---------------------------------------------------------------------------
__global__ void k_build_v(const float* __restrict__ X,
                          const float* __restrict__ theta,
                          const float* __restrict__ theta_links,
                          __bf16* __restrict__ Vbf) {
  int idx = blockIdx.x * blockDim.x + threadIdx.x;   // over 96*2048
  if (idx >= N_DH * N_LP) return;
  int dh = idx / N_LP, l = idx % N_LP;
  float v = 0.0f;
  if (l < N_L) {
    const float* x = X + ((size_t)dh * N_L + l) * 5 + 1;
    v = fmaf(x[0], theta[0],
        fmaf(x[1], theta[1],
        fmaf(x[2], theta[2],
        fmaf(x[3], theta[3], theta_links[l]))));
  }
  Vbf[idx] = (__bf16)v;
}

// ---------------------------------------------------------------------------
// Kernel 2: GEMM1  Vf(96 x 20000) = V(96 x 2000) @ D(2000 x 20000)
// One wave owns a 96x16 path strip (6 WMMA m-tiles), walks full K.
// D is read exactly once total; converted f32->bf16 in registers.
// B layout (contiguous K, matches A): lanes 0-15 supply K=0-15 of the step, lanes 16-31 K=16-31.
// ---------------------------------------------------------------------------
__global__ void k_gemm1(const __bf16* __restrict__ Vbf,
                        const float*  __restrict__ D,
                        float* __restrict__ Vf) {
  const int lane = threadIdx.x & 31;
  const int wave = blockIdx.x * (blockDim.x >> 5) + (threadIdx.x >> 5);
  if (wave >= N_P / 16) return;
  const int pbase = wave * 16;
  const int nn = lane & 15;   // path-in-tile (B/C column), also A row-in-tile
  const int kh = lane >> 4;   // K half selector

  f32x8 acc[6] = {};

  // Main K loop: 62 iterations of K=32 covering K < 1984; tail after.
  int k0 = 0;
  for (; k0 + 32 <= N_L; k0 += 32) {
    const float* dptr = D + (size_t)(k0 + kh * 16) * N_P + (pbase + nn);
    __builtin_prefetch((const void*)(dptr + (size_t)32 * N_P), 0, 1);
    bf16x16 b;
#pragma unroll
    for (int j = 0; j < 16; ++j)
      b[j] = (__bf16)dptr[(size_t)j * N_P];
#pragma unroll
    for (int m = 0; m < 6; ++m) {
      const bf16x16 a = load_a_frag(Vbf + (size_t)(m * 16 + nn) * N_LP, k0, kh);
      acc[m] = __builtin_amdgcn_wmma_f32_16x16x32_bf16(
                   false, a, false, b, (short)0, acc[m], false, false);
    }
  }
  // Tail: k0 = 1984, valid K = 1984..1999 (kh==0 half only); A is zero-padded.
  {
    bf16x16 b;
    if (kh == 0) {
      const float* dptr = D + (size_t)k0 * N_P + (pbase + nn);
#pragma unroll
      for (int j = 0; j < 16; ++j)
        b[j] = (__bf16)dptr[(size_t)j * N_P];
    } else {
#pragma unroll
      for (int j = 0; j < 16; ++j)
        b[j] = (__bf16)0.0f;
    }
#pragma unroll
    for (int m = 0; m < 6; ++m) {
      const bf16x16 a = load_a_frag(Vbf + (size_t)(m * 16 + nn) * N_LP, k0, kh);
      acc[m] = __builtin_amdgcn_wmma_f32_16x16x32_bf16(
                   false, a, false, b, (short)0, acc[m], false, false);
    }
  }

  // C layout: VGPR r holds M=r (lanes 0-15) / M=r+8 (lanes 16-31), N=lane&15.
#pragma unroll
  for (int m = 0; m < 6; ++m)
#pragma unroll
    for (int r = 0; r < 8; ++r)
      Vf[(size_t)(m * 16 + r + 8 * kh) * N_P + pbase + nn] = acc[m][r];
}

// ---------------------------------------------------------------------------
// Kernel 3: grouped softmax over contiguous 4-path OD groups, + psc bias,
// scaled by q = sqrt_q^2; emit f in bf16 (WMMA A operand for GEMM2).
// ---------------------------------------------------------------------------
__global__ void k_softmax(const float* __restrict__ Vf,
                          const float* __restrict__ sqrt_q,
                          const float* __restrict__ psf,
                          const float* __restrict__ pscf,
                          __bf16* __restrict__ fbf) {
  int idx = blockIdx.x * blockDim.x + threadIdx.x;   // over 96*5000
  if (idx >= N_DH * N_ODS) return;
  int dh = idx / N_ODS, od = idx % N_ODS;
  size_t base = (size_t)dh * N_P + (size_t)od * 4;
  float c = pscf[0];
  float v0 = Vf[base + 0] + c * __logf(psf[od * 4 + 0]);
  float v1 = Vf[base + 1] + c * __logf(psf[od * 4 + 1]);
  float v2 = Vf[base + 2] + c * __logf(psf[od * 4 + 2]);
  float v3 = Vf[base + 3] + c * __logf(psf[od * 4 + 3]);
  float vm = fmaxf(fmaxf(v0, v1), fmaxf(v2, v3));
  float e0 = __expf(v0 - vm), e1 = __expf(v1 - vm);
  float e2 = __expf(v2 - vm), e3 = __expf(v3 - vm);
  float q = sqrt_q[od]; q *= q;
  float inv = q / (e0 + e1 + e2 + e3);
  fbf[base + 0] = (__bf16)(e0 * inv);
  fbf[base + 1] = (__bf16)(e1 * inv);
  fbf[base + 2] = (__bf16)(e2 * inv);
  fbf[base + 3] = (__bf16)(e3 * inv);
}

// ---------------------------------------------------------------------------
// Kernel 4: GEMM2  part[c] = f(96 x 20000) @ D^T(20000 x 2000) k-chunk, split-K.
// B[k=p][n=l] = D[l][p]: each lane reads 64B contiguous from its link row.
// Deterministic: each (strip, chunk) writes its own partial buffer.
// ---------------------------------------------------------------------------
__global__ void k_gemm2(const __bf16* __restrict__ fbf,
                        const float*  __restrict__ D,
                        float* __restrict__ part) {
  const int lane = threadIdx.x & 31;
  const int gid  = blockIdx.x * (blockDim.x >> 5) + (threadIdx.x >> 5);
  if (gid >= (N_L / 16) * KSPLIT) return;
  const int strip = gid % (N_L / 16);
  const int chunk = gid / (N_L / 16);
  const int lbase = strip * 16;
  const int nn = lane & 15;   // link-in-tile
  const int kh = lane >> 4;

  const int ksteps = (N_P / 32) / KSPLIT;       // 125
  const int kbeg   = chunk * ksteps * 32;       // multiple of 32

  f32x8 acc[6] = {};

  for (int s = 0; s < ksteps; ++s) {
    const int k0 = kbeg + s * 32;
    const float* dptr = D + (size_t)(lbase + nn) * N_P + k0 + kh * 16;
    __builtin_prefetch((const void*)(dptr + 256), 0, 1);
    f32x4 q0 = *(const f32x4*)(dptr + 0);
    f32x4 q1 = *(const f32x4*)(dptr + 4);
    f32x4 q2 = *(const f32x4*)(dptr + 8);
    f32x4 q3 = *(const f32x4*)(dptr + 12);
    bf16x16 b;
#pragma unroll
    for (int j = 0; j < 4; ++j) {
      b[j + 0]  = (__bf16)q0[j];
      b[j + 4]  = (__bf16)q1[j];
      b[j + 8]  = (__bf16)q2[j];
      b[j + 12] = (__bf16)q3[j];
    }
#pragma unroll
    for (int m = 0; m < 6; ++m) {
      const bf16x16 a = load_a_frag(fbf + (size_t)(m * 16 + nn) * N_P, k0, kh);
      acc[m] = __builtin_amdgcn_wmma_f32_16x16x32_bf16(
                   false, a, false, b, (short)0, acc[m], false, false);
    }
  }

  float* pout = part + (size_t)chunk * N_DH * N_L;
#pragma unroll
  for (int m = 0; m < 6; ++m)
#pragma unroll
    for (int r = 0; r < 8; ++r)
      pout[(size_t)(m * 16 + r + 8 * kh) * N_L + lbase + nn] = acc[m][r];
}

// ---------------------------------------------------------------------------
// Kernel 5: sum split-K partials + ReLU -> d_out (96 x 2000 f32)
// ---------------------------------------------------------------------------
__global__ void k_reduce(const float* __restrict__ part,
                         float* __restrict__ out) {
  int idx = blockIdx.x * blockDim.x + threadIdx.x;   // over 96*2000
  if (idx >= N_DH * N_L) return;
  float s = 0.0f;
#pragma unroll
  for (int c = 0; c < KSPLIT; ++c)
    s += part[(size_t)c * N_DH * N_L + idx];
  out[idx] = fmaxf(s, 0.0f);
}

// ---------------------------------------------------------------------------
extern "C" void kernel_launch(void* const* d_in, const int* in_sizes, int n_in,
                              void* d_out, int out_size, void* d_ws, size_t ws_size,
                              hipStream_t stream) {
  (void)in_sizes; (void)n_in; (void)out_size; (void)ws_size;
  const float* X           = (const float*)d_in[0];
  const float* theta       = (const float*)d_in[1];
  const float* theta_links = (const float*)d_in[2];
  const float* sqrt_q      = (const float*)d_in[3];
  const float* psf         = (const float*)d_in[4];
  const float* pscf        = (const float*)d_in[5];
  const float* D           = (const float*)d_in[6];
  // d_in[7] (path_od) unused: groups are contiguous blocks of PATHS_PER_OD=4.
  float* out = (float*)d_out;

  char* ws = (char*)d_ws;
  __bf16* Vbf  = (__bf16*)(ws + OFF_VBF);
  __bf16* fbf  = (__bf16*)(ws + OFF_FBF);
  float*  Vf   = (float*) (ws + OFF_VF);
  float*  part = (float*) (ws + OFF_PART);

  k_build_v<<<(N_DH * N_LP + 255) / 256, 256, 0, stream>>>(X, theta, theta_links, Vbf);

  // 1250 waves, 8 waves (256 threads) per block
  k_gemm1<<<(N_P / 16 + 7) / 8, 256, 0, stream>>>(Vbf, D, Vf);

  k_softmax<<<(N_DH * N_ODS + 255) / 256, 256, 0, stream>>>(Vf, sqrt_q, psf, pscf, fbf);

  // 125 strips * KSPLIT chunks = 625 waves
  k_gemm2<<<((N_L / 16) * KSPLIT + 7) / 8, 256, 0, stream>>>(fbf, D, part);

  k_reduce<<<(N_DH * N_L + 255) / 256, 256, 0, stream>>>(part, out);
}
